// MLA_75625784148448
// MI455X (gfx1250) — matrix operand order repaired and checked by reading, not compiled, not run on hardware
//
#include <hip/hip_runtime.h>

// ---------------------------------------------------------------------------
// MLA forward for MI455X (gfx1250), wave32, WMMA bf16 path.
// B=2, S=2048, N_DIM=2048, H=16, Q_RANK=1536, KV_RANK=512,
// ROPE=64, NOPE=128, VD=128, QK_HEAD=192.
// ~212 GFLOP vs ~100MB HBM traffic -> compute bound; all matrix math through
// v_wmma_f32_16x16x32_bf16 (fp32 accumulate).
// GEMM: 2x2 register blocking per wave (16 FLOP/byte of fragment traffic).
// Attention: 4 waves/block share K/V tiles staged in LDS via
// global_load_async_to_lds_b128 (ASYNCcnt path, inline asm since the clang
// builtin requires HIP-inexpressible __device__/__shared__ AS pointers),
// double-buffered to overlap the async engine with WMMA compute.
// ---------------------------------------------------------------------------

typedef __attribute__((ext_vector_type(16))) __bf16   v16bf;
typedef __attribute__((ext_vector_type(8)))  float    v8f;
typedef __attribute__((ext_vector_type(4)))  unsigned u32x4;

#define B_SZ   2
#define S_LEN  2048
#define N_DIM  2048
#define HEADS  16
#define Q_RANK 1536
#define KV_RANK 512
#define ROPE_D 64
#define NOPE_D 128
#define VD_D   128
#define QKH_D  192               // NOPE + ROPE
#define KVW    (NOPE_D + VD_D)   // 256 per head in kv projection
#define ROWS   (B_SZ * S_LEN)    // 4096

#if defined(__gfx1250__)
#define HAVE_ASYNC_LDS 1
#else
#define HAVE_ASYNC_LDS 0
#endif

// Async global->LDS 16B copy. ISA: VDST = 32-bit LDS byte address,
// VADDR = 64-bit global address, SADDR = off (GV mode). Shared-memory flat
// apertures are 4GB-aligned, so addr[31:0] of a generic shared pointer is the
// LDS byte offset.
static __device__ __forceinline__ void async_cp16(void* dst, const void* src) {
#if HAVE_ASYNC_LDS
    unsigned lds = (unsigned)(unsigned long long)dst;
    asm volatile("global_load_async_to_lds_b128 %0, %1, off"
                 :: "v"(lds), "v"(src) : "memory");
#else
    *(u32x4*)dst = *(const u32x4*)src;
#endif
}

static __device__ __forceinline__ void wait_async() {
#if HAVE_ASYNC_LDS
#if __has_builtin(__builtin_amdgcn_s_wait_asynccnt)
    __builtin_amdgcn_s_wait_asynccnt(0);
#else
    asm volatile("s_wait_asynccnt 0x0" ::: "memory");
#endif
#endif
}

// ---- bf16 <-> f32 (bitwise, RNE) ----
static __device__ __forceinline__ __bf16 f2bf(float f) {
    union { float f; unsigned u; } a; a.f = f;
    unsigned r = a.u + 0x7FFFu + ((a.u >> 16) & 1u);
    union { unsigned short s; __bf16 b; } o; o.s = (unsigned short)(r >> 16);
    return o.b;
}
static __device__ __forceinline__ float bf2f(__bf16 b) {
    union { unsigned short s; __bf16 b; } i; i.b = b;
    union { unsigned u; float f; } o; o.u = ((unsigned)i.s) << 16;
    return o.f;
}

static __device__ __forceinline__ v8f wmma_bf16(v16bf a, v16bf b, v8f c) {
    return __builtin_amdgcn_wmma_f32_16x16x32_bf16(false, a, false, b,
                                                   (short)0, c, false, false);
}

// ---- 16-lane row reductions (rows live in 16-lane halves of the wave) ----
static __device__ __forceinline__ float rowmax16(float x) {
    x = fmaxf(x, __shfl_xor(x, 1, 32));
    x = fmaxf(x, __shfl_xor(x, 2, 32));
    x = fmaxf(x, __shfl_xor(x, 4, 32));
    x = fmaxf(x, __shfl_xor(x, 8, 32));
    return x;
}
static __device__ __forceinline__ float rowsum16(float x) {
    x += __shfl_xor(x, 1, 32);
    x += __shfl_xor(x, 2, 32);
    x += __shfl_xor(x, 4, 32);
    x += __shfl_xor(x, 8, 32);
    return x;
}

// ---------------------------------------------------------------------------
// fp32 -> bf16 conversion
// ---------------------------------------------------------------------------
__global__ void cvt_f32_bf16(const float* __restrict__ in,
                             __bf16* __restrict__ out, unsigned n) {
    unsigned i = blockIdx.x * 256u + threadIdx.x;
    if (i < n) out[i] = f2bf(in[i]);
}

// ---------------------------------------------------------------------------
// WMMA GEMM: C[M,N] = A[M,K] * Bt[N,K]^T (bf16 in, fp32 acc).
// grid = (N/64, M/64), block = 128 (4 waves, each a 32x32 register tile:
// 2 A-frags + 2 B-frags -> 4 wmma per k-step).
// ---------------------------------------------------------------------------
static __device__ __forceinline__ void store_tile(void* Cout, int ldc, int m,
                                                  int n, v8f acc, int c_is_f32,
                                                  int lo, int hi) {
    if (c_is_f32) {
        float* C = (float*)Cout;
#pragma unroll
        for (int v = 0; v < 8; ++v)
            C[(size_t)(m + v + 8 * hi) * ldc + n + lo] = acc[v];
    } else {
        __bf16* C = (__bf16*)Cout;
#pragma unroll
        for (int v = 0; v < 8; ++v)
            C[(size_t)(m + v + 8 * hi) * ldc + n + lo] = f2bf(acc[v]);
    }
}

__global__ __launch_bounds__(128)
void gemm_wmma_bf16(const __bf16* __restrict__ A, int lda,
                    const __bf16* __restrict__ Bt, int ldb,
                    void* __restrict__ Cout, int ldc,
                    int K, int c_is_f32) {
    const int lane = threadIdx.x & 31;
    const int w    = threadIdx.x >> 5;
    const int lo   = lane & 15;
    const int hi   = lane >> 4;
    const int m0   = blockIdx.y * 64 + (w >> 1) * 32;
    const int n0   = blockIdx.x * 64 + (w & 1) * 32;

    v8f acc00 = {}, acc01 = {}, acc10 = {}, acc11 = {};
    const __bf16* a0 = A + (size_t)(m0 + lo) * lda;
    const __bf16* a1 = A + (size_t)(m0 + 16 + lo) * lda;
    const __bf16* b0 = Bt + (size_t)(n0 + lo) * ldb + 16 * hi;
    const __bf16* b1 = Bt + (size_t)(n0 + 16 + lo) * ldb + 16 * hi;

    for (int k0 = 0; k0 < K; k0 += 32) {
        union { v16bf v; u32x4 q[2]; } uA0, uA1, uB0, uB1;
        // A fragment (16x32): lanes 0-15 K{0..15}, lanes 16-31 K{16..31}
        const u32x4* pa0 = (const u32x4*)(a0 + k0);
        uA0.q[0] = pa0[2 * hi];  uA0.q[1] = pa0[2 * hi + 1];
        const u32x4* pa1 = (const u32x4*)(a1 + k0);
        uA1.q[0] = pa1[2 * hi];  uA1.q[1] = pa1[2 * hi + 1];
        // B fragment (32x16): lane = column, 16 contiguous K values
        const u32x4* pb0 = (const u32x4*)(b0 + k0);
        uB0.q[0] = pb0[0];   uB0.q[1] = pb0[1];
        const u32x4* pb1 = (const u32x4*)(b1 + k0);
        uB1.q[0] = pb1[0];   uB1.q[1] = pb1[1];

        acc00 = wmma_bf16(uA0.v, uB0.v, acc00);
        acc01 = wmma_bf16(uA0.v, uB1.v, acc01);
        acc10 = wmma_bf16(uA1.v, uB0.v, acc10);
        acc11 = wmma_bf16(uA1.v, uB1.v, acc11);
    }

    store_tile(Cout, ldc, m0,      n0,      acc00, c_is_f32, lo, hi);
    store_tile(Cout, ldc, m0,      n0 + 16, acc01, c_is_f32, lo, hi);
    store_tile(Cout, ldc, m0 + 16, n0,      acc10, c_is_f32, lo, hi);
    store_tile(Cout, ldc, m0 + 16, n0 + 16, acc11, c_is_f32, lo, hi);
}

// ---------------------------------------------------------------------------
// RMSNorm in place over bf16 rows (fp32 math). One block per row.
// ---------------------------------------------------------------------------
__global__ __launch_bounds__(256)
void rmsnorm_bf16(__bf16* __restrict__ x, const float* __restrict__ w,
                  int W, int stride) {
    __shared__ float red[256];
    __bf16* p = x + (size_t)blockIdx.x * stride;
    float ss = 0.f;
    for (int i = threadIdx.x; i < W; i += 256) {
        float f = bf2f(p[i]);
        ss += f * f;
    }
    red[threadIdx.x] = ss;
    __syncthreads();
    for (int s = 128; s > 0; s >>= 1) {
        if ((int)threadIdx.x < s) red[threadIdx.x] += red[threadIdx.x + s];
        __syncthreads();
    }
    float inv = rsqrtf(red[0] / (float)W + 1e-6f);
    for (int i = threadIdx.x; i < W; i += 256) {
        float f = bf2f(p[i]);
        p[i] = f2bf(f * inv * w[i]);
    }
}

// ---------------------------------------------------------------------------
// RoPE kernels
// ---------------------------------------------------------------------------
__global__ __launch_bounds__(32)
void rope_q_kernel(__bf16* __restrict__ q, const float* __restrict__ fc) {
    int row = blockIdx.x, h = blockIdx.y;
    int s = row & (S_LEN - 1);
    int t = threadIdx.x;
    __bf16* p = q + (size_t)row * (HEADS * QKH_D) + h * QKH_D + NOPE_D + 2 * t;
    float c = fc[s * 64 + 2 * t], sn = fc[s * 64 + 2 * t + 1];
    float xr = bf2f(p[0]), xi = bf2f(p[1]);
    p[0] = f2bf(xr * c - xi * sn);
    p[1] = f2bf(xr * sn + xi * c);
}

__global__ __launch_bounds__(32)
void rope_k_kernel(__bf16* __restrict__ cbuf, const float* __restrict__ fc) {
    int row = blockIdx.x;
    int s = row & (S_LEN - 1);
    int t = threadIdx.x;
    __bf16* p = cbuf + (size_t)row * 576 + 512 + 2 * t;
    float c = fc[s * 64 + 2 * t], sn = fc[s * 64 + 2 * t + 1];
    float xr = bf2f(p[0]), xi = bf2f(p[1]);
    p[0] = f2bf(xr * c - xi * sn);
    p[1] = f2bf(xr * sn + xi * c);
}

// ---------------------------------------------------------------------------
// Flash attention, causal. Block = 4 waves = 64 q rows (wave w owns 16 rows).
// K tile (32x192) and V tile (32x128) staged in LDS by async copies,
// double-buffered: stage(jj+1) is issued before computing on buffer jj.
//   qf : [ROWS, H*192]  (nope | roped pe per head)
//   kvf: [ROWS, H*256]  (k_nope d<128 | v d>=128 per head)
//   cf : [ROWS, 576]    (roped k_pe at cols 512..575, shared across heads)
//   ao : [ROWS, H*128]
// ---------------------------------------------------------------------------
static __device__ __forceinline__ void stage_kv(int tid, int b, int h, int kb,
                                                const __bf16* __restrict__ kvf,
                                                const __bf16* __restrict__ cf,
                                                __bf16* kbuf, __bf16* vbuf) {
    // K tile: 32 rows x 192 d  (k_nope from kvf, k_pe from cf) = 768 x 16B
#pragma unroll
    for (int it = 0; it < 6; ++it) {
        int li  = it * 128 + tid;
        int row = li / 24;           // 24 chunks of 8 bf16 per row
        int c8  = li % 24;
        const __bf16* src = (c8 < 16)
            ? kvf + (size_t)(b * S_LEN + kb + row) * (HEADS * KVW) + h * KVW + c8 * 8
            : cf  + (size_t)(b * S_LEN + kb + row) * 576 + 512 + (c8 - 16) * 8;
        async_cp16(kbuf + row * 192 + c8 * 8, src);
    }
    // V tile: 32 rows x 128 d = 512 x 16B
#pragma unroll
    for (int it = 0; it < 4; ++it) {
        int li  = it * 128 + tid;
        int row = li / 16;
        int c8  = li % 16;
        const __bf16* src = kvf + (size_t)(b * S_LEN + kb + row) * (HEADS * KVW)
                                + h * KVW + NOPE_D + c8 * 8;
        async_cp16(vbuf + row * 128 + c8 * 8, src);
    }
}

__global__ __launch_bounds__(128)
void mla_attn_kernel(const __bf16* __restrict__ qf,
                     const __bf16* __restrict__ kvf,
                     const __bf16* __restrict__ cf,
                     __bf16* __restrict__ ao) {
    __shared__ __attribute__((aligned(32))) __bf16 kbuf[2][32 * 192];
    __shared__ __attribute__((aligned(32))) __bf16 vbuf[2][32 * 128];
    __shared__ __attribute__((aligned(32))) __bf16 pbuf[4][16 * 32];

    const int tid  = threadIdx.x;
    const int w    = tid >> 5;
    const int lane = tid & 31;
    const int lo = lane & 15, hi = lane >> 4;
    const int m0b = blockIdx.x * 64;     // block q-row base (seq index)
    const int m0  = m0b + w * 16;        // this wave's q-row tile
    const int h   = blockIdx.y;
    const int b   = blockIdx.z;

    // Q fragments: 6 chunks of 32 along d=192, A-matrix layout, in VGPRs
    v16bf qfrag[6];
    {
        const __bf16* qrow = qf + (size_t)(b * S_LEN + m0 + lo) * (HEADS * QKH_D)
                                + h * QKH_D;
#pragma unroll
        for (int c = 0; c < 6; ++c) {
            const u32x4* p = (const u32x4*)(qrow + 32 * c);
            union { v16bf v; u32x4 q[2]; } u;
            u.q[0] = p[2 * hi];
            u.q[1] = p[2 * hi + 1];
            qfrag[c] = u.v;
        }
    }

    v8f   o[8];
    float mr[8], lr[8];
#pragma unroll
    for (int t = 0; t < 8; ++t) o[t] = (v8f){};
#pragma unroll
    for (int v = 0; v < 8; ++v) { mr[v] = -3.0e38f; lr[v] = 0.0f; }

    const float scale = 0.07216878364870323f;   // 1/sqrt(192)
    const int nblk = (m0b + 63) / 32 + 1;       // cover k cols 0..m0b+63

    stage_kv(tid, b, h, 0, kvf, cf, kbuf[0], vbuf[0]);

    for (int jj = 0; jj < nblk; ++jj) {
        const int kb  = jj * 32;
        const int cur = jj & 1;

        wait_async();          // our staged loads for buffer `cur` are done
        __syncthreads();       // everyone's are done; prev compute finished

        if (jj + 1 < nblk)     // prefetch next tile while we compute this one
            stage_kv(tid, b, h, kb + 32, kvf, cf, kbuf[cur ^ 1], vbuf[cur ^ 1]);

        const bool active = (kb <= m0 + 15);    // causal: wave has live cols
        if (active) {
            // ---- S = scale * Q K^T over 32 k columns (two 16x16 tiles) ----
            v8f s0 = {}, s1 = {};
#pragma unroll
            for (int c = 0; c < 6; ++c) {
                const u32x4* p0 =
                    (const u32x4*)(&kbuf[cur][lo * 192 + 32 * c + 16 * hi]);
                union { v16bf v; u32x4 q[2]; } u0;
                u0.q[0] = p0[0]; u0.q[1] = p0[1];
                s0 = wmma_bf16(qfrag[c], u0.v, s0);

                const u32x4* p1 =
                    (const u32x4*)(&kbuf[cur][(16 + lo) * 192 + 32 * c + 16 * hi]);
                union { v16bf v; u32x4 q[2]; } u1;
                u1.q[0] = p1[0]; u1.q[1] = p1[1];
                s1 = wmma_bf16(qfrag[c], u1.v, s1);
            }

            // ---- online softmax + stage P ----
            const int qbase = m0 + 8 * hi;
#pragma unroll
            for (int v = 0; v < 8; ++v) {
                float a0 = s0[v] * scale;
                float a1 = s1[v] * scale;
                if (kb + lo      > qbase + v) a0 = -3.0e38f;
                if (kb + 16 + lo > qbase + v) a1 = -3.0e38f;
                float mx    = rowmax16(fmaxf(a0, a1));
                float mnew  = fmaxf(mr[v], mx);
                float alpha = __expf(mr[v] - mnew);
                float p0 = __expf(a0 - mnew);
                float p1 = __expf(a1 - mnew);
                float rs = rowsum16(p0 + p1);
                lr[v] = lr[v] * alpha + rs;
                mr[v] = mnew;
#pragma unroll
                for (int t = 0; t < 8; ++t) o[t][v] *= alpha;
                pbuf[w][(v + 8 * hi) * 32 + lo]      = f2bf(p0);
                pbuf[w][(v + 8 * hi) * 32 + 16 + lo] = f2bf(p1);
            }
        }
        __syncthreads();   // P visible; K reads done before next overwrite

        if (active) {
            // ---- reload P as A-fragment (16x32) ----
            v16bf pfrag;
            {
                const u32x4* p = (const u32x4*)(&pbuf[w][lo * 32]);
                union { v16bf v; u32x4 q[2]; } u;
                u.q[0] = p[2 * hi];
                u.q[1] = p[2 * hi + 1];
                pfrag = u.v;
            }
            // ---- O += P * V  (8 column tiles of 16) ----
#pragma unroll
            for (int t = 0; t < 8; ++t) {
                v16bf vfrag;
#pragma unroll
                for (int v2 = 0; v2 < 8; ++v2) {
                    const int kr = 2 * v2 + 16 * hi;
                    vfrag[2 * v2]     = vbuf[cur][kr * 128 + 16 * t + lo];
                    vfrag[2 * v2 + 1] = vbuf[cur][(kr + 1) * 128 + 16 * t + lo];
                }
                o[t] = wmma_bf16(pfrag, vfrag, o[t]);
            }
        }
        __syncthreads();   // V reads done before next overwrite
    }

    // ---- normalize and store [ROWS, H*128] ----
#pragma unroll
    for (int t = 0; t < 8; ++t) {
#pragma unroll
        for (int v = 0; v < 8; ++v) {
            const int row = b * S_LEN + m0 + v + 8 * hi;
            ao[(size_t)row * (HEADS * VD_D) + h * VD_D + 16 * t + lo] =
                f2bf(o[t][v] / lr[v]);
        }
    }
}

// ---------------------------------------------------------------------------
// Host-side orchestration
// ---------------------------------------------------------------------------
extern "C" void kernel_launch(void* const* d_in, const int* in_sizes, int n_in,
                              void* d_out, int out_size, void* d_ws, size_t ws_size,
                              hipStream_t stream) {
    const float* x        = (const float*)d_in[0];
    const float* freqs    = (const float*)d_in[1];
    const float* wq_a     = (const float*)d_in[2];
    const float* q_norm_w = (const float*)d_in[3];
    const float* wq_b     = (const float*)d_in[4];
    const float* wkv_a    = (const float*)d_in[5];
    const float* kv_norm_w= (const float*)d_in[6];
    const float* wkv_b    = (const float*)d_in[7];
    const float* wo       = (const float*)d_in[8];
    float* out = (float*)d_out;

    __bf16* ws = (__bf16*)d_ws;
    const size_t N_X    = (size_t)ROWS * N_DIM;
    const size_t N_WQA  = (size_t)Q_RANK * N_DIM;
    const size_t N_WQB  = (size_t)HEADS * QKH_D * Q_RANK;
    const size_t N_WKVA = (size_t)(KV_RANK + ROPE_D) * N_DIM;
    const size_t N_WKVB = (size_t)HEADS * KVW * KV_RANK;
    const size_t N_WO   = (size_t)N_DIM * HEADS * VD_D;
    const size_t N_QLAT = (size_t)ROWS * Q_RANK;
    const size_t N_QF   = (size_t)ROWS * HEADS * QKH_D;
    const size_t N_CF   = (size_t)ROWS * 576;
    const size_t N_KVF  = (size_t)ROWS * HEADS * KVW;
    const size_t N_AO   = (size_t)ROWS * HEADS * VD_D;

    __bf16* x_bf    = ws;  ws += N_X;
    __bf16* wqa_bf  = ws;  ws += N_WQA;
    __bf16* wqb_bf  = ws;  ws += N_WQB;
    __bf16* wkva_bf = ws;  ws += N_WKVA;
    __bf16* wkvb_bf = ws;  ws += N_WKVB;
    __bf16* wo_bf   = ws;  ws += N_WO;
    __bf16* qlat_bf = ws;  ws += N_QLAT;
    __bf16* qf_bf   = ws;  ws += N_QF;
    __bf16* cf_bf   = ws;  ws += N_CF;
    __bf16* kvf_bf  = ws;  ws += N_KVF;
    __bf16* ao_bf   = ws;  ws += N_AO;

    (void)in_sizes; (void)n_in; (void)out_size; (void)ws_size;

    auto cvt = [&](const float* src, __bf16* dst, size_t n) {
        cvt_f32_bf16<<<(unsigned)((n + 255) / 256), 256, 0, stream>>>(
            src, dst, (unsigned)n);
    };
    cvt(x, x_bf, N_X);
    cvt(wq_a, wqa_bf, N_WQA);
    cvt(wq_b, wqb_bf, N_WQB);
    cvt(wkv_a, wkva_bf, N_WKVA);
    cvt(wkv_b, wkvb_bf, N_WKVB);
    cvt(wo, wo_bf, N_WO);

    // q_lat = x @ wq_a.T   [4096,1536]
    gemm_wmma_bf16<<<dim3(Q_RANK / 64, ROWS / 64), 128, 0, stream>>>(
        x_bf, N_DIM, wqa_bf, N_DIM, qlat_bf, Q_RANK, N_DIM, 0);
    rmsnorm_bf16<<<ROWS, 256, 0, stream>>>(qlat_bf, q_norm_w, Q_RANK, Q_RANK);
    // q = q_lat @ wq_b.T   [4096,3072]
    gemm_wmma_bf16<<<dim3((HEADS * QKH_D) / 64, ROWS / 64), 128, 0, stream>>>(
        qlat_bf, Q_RANK, wqb_bf, Q_RANK, qf_bf, HEADS * QKH_D, Q_RANK, 0);
    // c = x @ wkv_a.T      [4096,576]
    gemm_wmma_bf16<<<dim3(576 / 64, ROWS / 64), 128, 0, stream>>>(
        x_bf, N_DIM, wkva_bf, N_DIM, cf_bf, 576, N_DIM, 0);
    rmsnorm_bf16<<<ROWS, 256, 0, stream>>>(cf_bf, kv_norm_w, KV_RANK, 576);
    rope_q_kernel<<<dim3(ROWS, HEADS), 32, 0, stream>>>(qf_bf, freqs);
    rope_k_kernel<<<ROWS, 32, 0, stream>>>(cf_bf, freqs);
    // kv = rmsnorm(kv_lat) @ wkv_b.T   [4096,4096]
    gemm_wmma_bf16<<<dim3((HEADS * KVW) / 64, ROWS / 64), 128, 0, stream>>>(
        cf_bf, 576, wkvb_bf, KV_RANK, kvf_bf, HEADS * KVW, KV_RANK, 0);
    // causal flash attention: block = 64 q rows x one head x one batch
    mla_attn_kernel<<<dim3(S_LEN / 64, HEADS, B_SZ), 128, 0, stream>>>(
        qf_bf, kvf_bf, cf_bf, ao_bf);
    // out = attn_out @ wo.T  (fp32 result)
    gemm_wmma_bf16<<<dim3(N_DIM / 64, ROWS / 64), 128, 0, stream>>>(
        ao_bf, HEADS * VD_D, wo_bf, HEADS * VD_D, out, N_DIM, HEADS * VD_D, 1);
}
